// MaskedAttentionDecoder_42425686949963
// MI455X (gfx1250) — matrix operand order repaired and checked by reading, not compiled, not run on hardware
//
#include <hip/hip_runtime.h>
#include <hip/hip_bf16.h>

// MI455X / gfx1250, wave32. All matrix math via v_wmma_f32_16x16x32_bf16.
// Network: 9-layer masked-attention decoder, D=256, H=8 (HD=32), Q=100, B=8.
// Roofline: dominated by streaming mask_feat (134MB per _pred, 10 preds) and
// level-2 K/V projections; bf16 WMMA with f32 accumulate matches fp32 ref
// within bf16 rounding. GEMM: 8 waves/block, block tile 32x256, wave tile
// 16x64 -> 4 WMMAs per K-step with A-fragment reuse; B next-tile prefetch
// (global_prefetch_b8) to exploit the 192MB L2.

#define QN   100
#define DN   256
#define HN   8
#define HD   32
#define BN   8
#define KCLS 21
#define MFHW 128

typedef __attribute__((ext_vector_type(16))) __bf16 v16bf;
typedef __attribute__((ext_vector_type(8)))  float  v8f;

// ---------------------------------------------------------------------------
// Generic batched GEMM:  C = act(A @ B + bias) [+ Res]
//   A: M x K   element(m,k) = transA ? A[k*lda+m] : A[m*lda+k]
//   B: K x N   element(k,n) = transB ? B[n*ldb+k] : B[k*ldb+n]
//   act: 0 none, 1 relu, 2 exact gelu
// Block = 256 threads (8 waves, 2Mx4N); block tile 32x256; wave tile 16x64.
// ---------------------------------------------------------------------------
__global__ void __launch_bounds__(256)
gemm_wmma(const float* __restrict__ A, const float* __restrict__ Bm,
          const float* __restrict__ bias, const float* __restrict__ Res,
          float* __restrict__ C,
          int M, int N, int Kd,
          int lda, int ldb, int ldc,
          long strideA, long strideB, long strideC,
          int transA, int transB, int act)
{
    int batch = blockIdx.z;
    A  += (long)batch * strideA;
    Bm += (long)batch * strideB;
    C  += (long)batch * strideC;
    const float* R = Res ? Res + (long)batch * strideC : nullptr;

    __shared__ float As[32 * 33];   // padded: conflict-free column gather
    __shared__ float Bs[32 * 256];

    int m0   = blockIdx.y * 32;
    int n0   = blockIdx.x * 256;
    int tid  = threadIdx.x;
    int wave = tid >> 5;            // 0..7
    int lane = tid & 31;
    int mw   = wave >> 2;           // 0..1  (M sub-tile)
    int nw   = wave & 3;            // 0..3  (N strip of 64)

    int mrow = (lane & 15) + mw * 16;  // A-frag row within block tile
    int ksel = (lane >> 4) ? 8 : 0;    // A-frag K-half select
    int ncol = lane & 15;              // B-frag / C col within 16
    int kb16 = (lane >> 4) ? 16 : 0;   // B-frag K base
    int rb   = (lane >> 4) ? 8 : 0;    // C-frag row base

    v8f acc0 = {}, acc1 = {}, acc2 = {}, acc3 = {};

    for (int k0 = 0; k0 < Kd; k0 += 32) {
        __syncthreads();   // previous iteration's LDS reads complete
        // stage A tile (32 x 32)
        for (int idx = tid; idx < 32 * 32; idx += 256) {
            int m, k;
            if (transA) { k = idx >> 5; m = idx & 31; }
            else        { m = idx >> 5; k = idx & 31; }
            int gm = m0 + m, gk = k0 + k;
            float v = 0.f;
            if (gm < M)
                v = transA ? A[(long)gk * lda + gm] : A[(long)gm * lda + gk];
            As[m * 33 + k] = v;
        }
        // stage B tile (32 x 256) as (k, n); prefetch next K-tile of B
        for (int idx = tid; idx < 32 * 256; idx += 256) {
            int k, n;
            if (transB) { n = idx >> 5; k = idx & 31; }
            else        { k = idx >> 8; n = idx & 255; }
            int gk = k0 + k, gn = n0 + n;
            float v = 0.f;
            if (gn < N) {
                const float* p = transB ? &Bm[(long)gn * ldb + gk]
                                        : &Bm[(long)gk * ldb + gn];
                v = *p;
                if (k0 + 32 < Kd)
                    __builtin_prefetch(p + (transB ? 32 : (long)32 * ldb), 0, 3);
            }
            Bs[k * 256 + n] = v;
        }
        __syncthreads();

        // A fragment: lanes 0-15 row m K{0..15}; lanes 16-31 K{16..31}
        v16bf af;
#pragma unroll
        for (int i = 0; i < 16; ++i) {
            int kk = i + kb16;
            af[i] = (__bf16)As[mrow * 33 + kk];
        }
        // 4 B fragments (N strips of 16) -> 4 WMMAs reusing af
#pragma unroll
        for (int j = 0; j < 4; ++j) {
            v16bf bfb;
            int col = nw * 64 + j * 16 + ncol;
#pragma unroll
            for (int i = 0; i < 16; ++i)
                bfb[i] = (__bf16)Bs[(kb16 + i) * 256 + col];
            v8f* accp = (j == 0) ? &acc0 : (j == 1) ? &acc1 : (j == 2) ? &acc2 : &acc3;
            *accp = __builtin_amdgcn_wmma_f32_16x16x32_bf16(false, af, false, bfb,
                                                            (short)0, *accp, false, false);
        }
    }

#pragma unroll
    for (int j = 0; j < 4; ++j) {
        const v8f& acc = (j == 0) ? acc0 : (j == 1) ? acc1 : (j == 2) ? acc2 : acc3;
        int gn = n0 + nw * 64 + j * 16 + ncol;
        if (gn < N) {
            float bv = bias ? bias[gn] : 0.f;
#pragma unroll
            for (int r = 0; r < 8; ++r) {
                int gm = m0 + mw * 16 + r + rb;
                if (gm < M) {
                    float v = acc[r] + bv;
                    if (act == 1)      v = fmaxf(v, 0.f);
                    else if (act == 2) v = 0.5f * v * (1.f + erff(v * 0.70710678118f));
                    if (R) v += R[(long)gm * ldc + gn];
                    C[(long)gm * ldc + gn] = v;
                }
            }
        }
    }
}

// ---------------------------------------------------------------------------
// Flash attention (online softmax), one wave per (b, h, 16-query tile).
// HD=32 == WMMA K, so QK^T and P.V are single-K-step WMMAs.
// bias: [B,Q,S] additive (-1e9/0) or nullptr (self-attn).
// ---------------------------------------------------------------------------
__global__ void attn_wmma(const float* __restrict__ qp, const float* __restrict__ kp,
                          const float* __restrict__ vp, const float* __restrict__ bias,
                          float* __restrict__ o, int S)
{
    int m0   = blockIdx.x * 16;
    int h    = blockIdx.y;
    int b    = blockIdx.z;
    int lane = threadIdx.x;

    const float scale = 0.17677669529663689f; // 1/sqrt(32)
    const float* qb = qp + (long)b * QN * DN + h * HD;
    const float* kb = kp + (long)b * S  * DN + h * HD;
    const float* vb = vp + (long)b * S  * DN + h * HD;
    const float* bb = bias ? bias + (long)b * QN * S : nullptr;

    __shared__ float sc[16][33];
    __shared__ float sfac[16];
    __shared__ float ssum[16];

    int mrow = lane & 15;
    int ksel = (lane >> 4) ? 8 : 0;
    int ncol = lane & 15;
    int kb16 = (lane >> 4) ? 16 : 0;
    int rb   = (lane >> 4) ? 8 : 0;
    int gm   = m0 + mrow;

    // Q A-fragment (scale folded in)
    v16bf aq;
#pragma unroll
    for (int i = 0; i < 16; ++i) {
        int kk = i + kb16;
        float v = (gm < QN) ? qb[(long)gm * DN + kk] * scale : 0.f;
        aq[i] = (__bf16)v;
    }

    v8f acc0 = {}, acc1 = {};
    float rmax = -1e30f, rsum = 0.f;   // row stats owned by lanes 0..15

    for (int s0 = 0; s0 < S; s0 += 32) {
        // K B-fragments for keys [s0, s0+16) and [s0+16, s0+32)
        v16bf bk0, bk1;
        int key0 = s0 + ncol, key1 = s0 + 16 + ncol;
#pragma unroll
        for (int i = 0; i < 16; ++i) {
            bk0[i] = (__bf16)((key0 < S) ? kb[(long)key0 * DN + kb16 + i] : 0.f);
            bk1[i] = (__bf16)((key1 < S) ? kb[(long)key1 * DN + kb16 + i] : 0.f);
        }
        v8f z = {};
        v8f sf0 = __builtin_amdgcn_wmma_f32_16x16x32_bf16(false, aq, false, bk0,
                                                          (short)0, z, false, false);
        v8f sf1 = __builtin_amdgcn_wmma_f32_16x16x32_bf16(false, aq, false, bk1,
                                                          (short)0, z, false, false);
        // scores -> LDS with bias & key bounds
#pragma unroll
        for (int r = 0; r < 8; ++r) {
            int row = r + rb;
            int qg  = m0 + row;
            float v0 = sf0[r], v1 = sf1[r];
            if (bb && qg < QN) {
                if (key0 < S) v0 += bb[(long)qg * S + key0];
                if (key1 < S) v1 += bb[(long)qg * S + key1];
            }
            if (key0 >= S) v0 = -1e30f;
            if (key1 >= S) v1 = -1e30f;
            sc[row][ncol]      = v0;
            sc[row][16 + ncol] = v1;
        }
        __syncthreads();

        // online-softmax update (lanes 0..15 each own one query row)
        if (lane < 16) {
            float cm = -1e30f;
#pragma unroll 4
            for (int j = 0; j < 32; ++j) cm = fmaxf(cm, sc[lane][j]);
            float nm  = fmaxf(rmax, cm);
            float fac = __expf(rmax - nm);
            float ps  = 0.f;
#pragma unroll 4
            for (int j = 0; j < 32; ++j) {
                float e = __expf(sc[lane][j] - nm);
                sc[lane][j] = e;
                ps += e;
            }
            rsum = rsum * fac + ps;
            rmax = nm;
            sfac[lane] = fac;
        }
        __syncthreads();

        // rescale output accumulators
#pragma unroll
        for (int r = 0; r < 8; ++r) {
            float f = sfac[r + rb];
            acc0[r] *= f;
            acc1[r] *= f;
        }
        // P A-fragment from LDS
        v16bf pf;
#pragma unroll
        for (int i = 0; i < 16; ++i) {
            int kk = i + kb16;
            pf[i] = (__bf16)sc[mrow][kk];
        }
        // V B-fragments (dims 0..15 / 16..31)
        v16bf bv0, bv1;
#pragma unroll
        for (int i = 0; i < 16; ++i) {
            int key = s0 + kb16 + i;
            float f0 = 0.f, f1 = 0.f;
            if (key < S) {
                f0 = vb[(long)key * DN + ncol];
                f1 = vb[(long)key * DN + 16 + ncol];
            }
            bv0[i] = (__bf16)f0;
            bv1[i] = (__bf16)f1;
        }
        acc0 = __builtin_amdgcn_wmma_f32_16x16x32_bf16(false, pf, false, bv0,
                                                       (short)0, acc0, false, false);
        acc1 = __builtin_amdgcn_wmma_f32_16x16x32_bf16(false, pf, false, bv1,
                                                       (short)0, acc1, false, false);
        __syncthreads();
    }

    if (lane < 16) ssum[lane] = rsum;
    __syncthreads();
#pragma unroll
    for (int r = 0; r < 8; ++r) {
        int qg = m0 + r + rb;
        if (qg < QN) {
            float inv = 1.f / ssum[r + rb];
            float* op = o + (long)(b * QN + qg) * DN + h * HD;
            op[ncol]      = acc0[r] * inv;
            op[16 + ncol] = acc1[r] * inv;
        }
    }
}

// ---------------------------------------------------------------------------
// LayerNorm over last dim (256). One block (256 threads) per row.
// ---------------------------------------------------------------------------
__global__ void layernorm_k(const float* __restrict__ x, const float* __restrict__ w,
                            const float* __restrict__ b, float* __restrict__ y)
{
    int row = blockIdx.x;
    int t   = threadIdx.x;
    float v = x[(long)row * DN + t];
    __shared__ float red[DN];
    red[t] = v;
    __syncthreads();
    for (int s = 128; s > 0; s >>= 1) {
        if (t < s) red[t] += red[t + s];
        __syncthreads();
    }
    float mean = red[0] * (1.f / DN);
    __syncthreads();
    float d = v - mean;
    red[t] = d * d;
    __syncthreads();
    for (int s = 128; s > 0; s >>= 1) {
        if (t < s) red[t] += red[t + s];
        __syncthreads();
    }
    float var = red[0] * (1.f / DN);
    y[(long)row * DN + t] = d * rsqrtf(var + 1e-5f) * w[t] + b[t];
}

// ---------------------------------------------------------------------------
// Attention-mask bias: antialiased (triangle) downsample of masks [128x128]
// to (h,w), am = sigmoid(x)<0.5 <=> x<0 -> bias -1e9; if a row masks every
// key, allow all. One block per (b,q) row; dynamic LDS = S floats.
// ---------------------------------------------------------------------------
__global__ void mask_bias_k(const float* __restrict__ masks, float* __restrict__ bias,
                            int h, int w)
{
    int bq = blockIdx.x;
    const float* mp = masks + (long)bq * (MFHW * MFHW);
    int S = h * w;
    float* bp = bias + (long)bq * S;

    extern __shared__ float sm[];
    __shared__ int anyAllowed;
    if (threadIdx.x == 0) anyAllowed = 0;

    float sy = (float)MFHW / h, sx = (float)MFHW / w;
    for (int s = threadIdx.x; s < S; s += blockDim.x) {
        int oy = s / w, ox = s - oy * w;
        float cy = (oy + 0.5f) * sy - 0.5f;
        float cx = (ox + 0.5f) * sx - 0.5f;
        int y0 = (int)ceilf(cy - sy), y1 = (int)floorf(cy + sy);
        int x0 = (int)ceilf(cx - sx), x1 = (int)floorf(cx + sx);
        float acc = 0.f, wsum = 0.f;
        for (int yy = y0; yy <= y1; ++yy) {
            float wy = 1.f - fabsf((float)yy - cy) / sy;
            if (wy <= 0.f) continue;
            int yc = min(max(yy, 0), MFHW - 1);
            for (int xx = x0; xx <= x1; ++xx) {
                float wx = 1.f - fabsf((float)xx - cx) / sx;
                if (wx <= 0.f) continue;
                int xc = min(max(xx, 0), MFHW - 1);
                float g = wy * wx;
                acc  += g * mp[yc * MFHW + xc];
                wsum += g;
            }
        }
        float v = acc / wsum;
        sm[s] = (v < 0.f) ? -1e9f : 0.f;
    }
    __syncthreads();
    int loc = 0;
    for (int s = threadIdx.x; s < S; s += blockDim.x)
        if (sm[s] == 0.f) loc = 1;
    if (loc) atomicOr(&anyAllowed, 1);
    __syncthreads();
    int allow = anyAllowed;
    for (int s = threadIdx.x; s < S; s += blockDim.x)
        bp[s] = allow ? sm[s] : 0.f;
}

// ---------------------------------------------------------------------------
extern "C" void kernel_launch(void* const* d_in, const int* in_sizes, int n_in,
                              void* d_out, int out_size, void* d_ws, size_t ws_size,
                              hipStream_t stream)
{
    (void)in_sizes; (void)n_in; (void)out_size; (void)ws_size;
    const float* q_in      = (const float*)d_in[0];
    const float* mask_feat = (const float*)d_in[1];
    const float* lvl[3]    = {(const float*)d_in[2], (const float*)d_in[3], (const float*)d_in[4]};
    // params in dict order
    const float* ln1_w = (const float*)d_in[5];
    const float* ln1_b = (const float*)d_in[6];
    const float* ca_in_w = (const float*)d_in[7];
    const float* ca_in_b = (const float*)d_in[8];
    const float* ca_out_w = (const float*)d_in[9];
    const float* ca_out_b = (const float*)d_in[10];
    const float* ln2_w = (const float*)d_in[11];
    const float* ln2_b = (const float*)d_in[12];
    const float* sa_in_w = (const float*)d_in[13];
    const float* sa_in_b = (const float*)d_in[14];
    const float* sa_out_w = (const float*)d_in[15];
    const float* sa_out_b = (const float*)d_in[16];
    const float* ln3_w = (const float*)d_in[17];
    const float* ln3_b = (const float*)d_in[18];
    const float* ff_w1 = (const float*)d_in[19];
    const float* ff_b1 = (const float*)d_in[20];
    const float* ff_w2 = (const float*)d_in[21];
    const float* ff_b2 = (const float*)d_in[22];
    const float* cls_w = (const float*)d_in[23];
    const float* cls_b = (const float*)d_in[24];
    const float* me_w1 = (const float*)d_in[25];
    const float* me_b1 = (const float*)d_in[26];
    const float* me_w2 = (const float*)d_in[27];
    const float* me_b2 = (const float*)d_in[28];
    const float* me_w3 = (const float*)d_in[29];
    const float* me_b3 = (const float*)d_in[30];

    float* out_logits = (float*)d_out;                     // [B,Q,21]
    float* out_masks  = (float*)d_out + BN * QN * KCLS;    // [B,Q,128,128] (also scratch)

    float* ws    = (float*)d_ws;
    float* qbuf  = ws;                 // [800,256]
    float* hq    = ws + 204800;        // [800,256]
    float* qp    = ws + 409600;        // [800,256]
    float* ob    = ws + 614400;        // [800,256]
    float* emb_a = ws + 819200;        // [800,256]
    float* emb_b = ws + 1024000;       // [800,256]
    float* ff1   = ws + 1228800;       // [800,1024]
    float* kpb   = ws + 2048000;       // [8,4096,256] max
    float* vpb   = ws + 10436608;      // [8,4096,256] max
    float* biasb = ws + 18825216;      // [8,100,4096] max

    hipMemcpyAsync(qbuf, q_in, (size_t)BN * QN * DN * sizeof(float),
                   hipMemcpyDeviceToDevice, stream);

    auto gemm = [&](const float* A, const float* Bm, const float* bias, const float* Res,
                    float* C, int M, int N, int K, int lda, int ldb, int ldc,
                    long sA, long sB, long sC, int tA, int tB, int act, int batches) {
        dim3 g((N + 255) / 256, (M + 31) / 32, batches);
        gemm_wmma<<<g, 256, 0, stream>>>(A, Bm, bias, Res, C, M, N, K,
                                         lda, ldb, ldc, sA, sB, sC, tA, tB, act);
    };

    auto pred = [&]() {
        // logits = q @ cls_w.T + cls_b
        gemm(qbuf, cls_w, cls_b, nullptr, out_logits, BN * QN, KCLS, DN,
             DN, DN, KCLS, 0, 0, 0, 0, 1, 0, 1);
        // mask embed MLP
        gemm(qbuf,  me_w1, me_b1, nullptr, emb_a, BN * QN, DN, DN, DN, DN, DN, 0, 0, 0, 0, 1, 1, 1);
        gemm(emb_a, me_w2, me_b2, nullptr, emb_b, BN * QN, DN, DN, DN, DN, DN, 0, 0, 0, 0, 1, 1, 1);
        gemm(emb_b, me_w3, me_b3, nullptr, emb_a, BN * QN, DN, DN, DN, DN, DN, 0, 0, 0, 0, 1, 0, 1);
        // masks[b] = m[b] (100x256) @ mask_feat[b] (256x16384)
        gemm(emb_a, mask_feat, nullptr, nullptr, out_masks, QN, MFHW * MFHW, DN,
             DN, MFHW * MFHW, MFHW * MFHW,
             (long)QN * DN, (long)DN * MFHW * MFHW, (long)QN * MFHW * MFHW,
             0, 0, 0, BN);
    };

    pred(); // init_masks

    const int HWsz[3] = {16, 32, 64};
    for (int i = 0; i < 9; ++i) {
        int hs = HWsz[i % 3];
        int S  = hs * hs;

        // attention bias from previous masks
        mask_bias_k<<<BN * QN, 256, S * sizeof(float), stream>>>(out_masks, biasb, hs, hs);

        // ---- masked cross-attention ----
        layernorm_k<<<BN * QN, DN, 0, stream>>>(qbuf, ln1_w + i * DN, ln1_b + i * DN, hq);
        const float* caw = ca_in_w + (long)i * 3 * DN * DN;
        const float* cab = ca_in_b + (long)i * 3 * DN;
        gemm(hq, caw, cab, nullptr, qp, BN * QN, DN, DN, DN, DN, DN, 0, 0, 0, 0, 1, 0, 1);
        // memory[b,s,d] = lvl[b,d,s]  -> transA with lda = S
        gemm(lvl[i % 3], caw + DN * DN,     cab + DN,     nullptr, kpb, S, DN, DN,
             S, DN, DN, (long)DN * S, 0, (long)S * DN, 1, 1, 0, BN);
        gemm(lvl[i % 3], caw + 2 * DN * DN, cab + 2 * DN, nullptr, vpb, S, DN, DN,
             S, DN, DN, (long)DN * S, 0, (long)S * DN, 1, 1, 0, BN);
        attn_wmma<<<dim3((QN + 15) / 16, HN, BN), 32, 0, stream>>>(qp, kpb, vpb, biasb, ob, S);
        gemm(ob, ca_out_w + (long)i * DN * DN, ca_out_b + i * DN, qbuf, qbuf,
             BN * QN, DN, DN, DN, DN, DN, 0, 0, 0, 0, 1, 0, 1);

        // ---- self-attention ----
        layernorm_k<<<BN * QN, DN, 0, stream>>>(qbuf, ln2_w + i * DN, ln2_b + i * DN, hq);
        const float* saw = sa_in_w + (long)i * 3 * DN * DN;
        const float* sab = sa_in_b + (long)i * 3 * DN;
        gemm(hq, saw,               sab,           nullptr, qp,  BN * QN, DN, DN, DN, DN, DN, 0, 0, 0, 0, 1, 0, 1);
        gemm(hq, saw + DN * DN,     sab + DN,      nullptr, kpb, BN * QN, DN, DN, DN, DN, DN, 0, 0, 0, 0, 1, 0, 1);
        gemm(hq, saw + 2 * DN * DN, sab + 2 * DN,  nullptr, vpb, BN * QN, DN, DN, DN, DN, DN, 0, 0, 0, 0, 1, 0, 1);
        attn_wmma<<<dim3((QN + 15) / 16, HN, BN), 32, 0, stream>>>(qp, kpb, vpb, nullptr, ob, QN);
        gemm(ob, sa_out_w + (long)i * DN * DN, sa_out_b + i * DN, qbuf, qbuf,
             BN * QN, DN, DN, DN, DN, DN, 0, 0, 0, 0, 1, 0, 1);

        // ---- FFN ----
        layernorm_k<<<BN * QN, DN, 0, stream>>>(qbuf, ln3_w + i * DN, ln3_b + i * DN, hq);
        gemm(hq, ff_w1 + (long)i * 4 * DN * DN, ff_b1 + (long)i * 4 * DN, nullptr, ff1,
             BN * QN, 4 * DN, DN, DN, DN, 4 * DN, 0, 0, 0, 0, 1, 2, 1);
        gemm(ff1, ff_w2 + (long)i * 4 * DN * DN, ff_b2 + (long)i * DN, qbuf, qbuf,
             BN * QN, DN, 4 * DN, 4 * DN, 4 * DN, DN, 0, 0, 0, 0, 1, 0, 1);

        pred();
    }
}
